// RNN_91070486544574
// MI455X (gfx1250) — compile-verified, matrix operand order checked by reading, not run on hardware
//
#include <hip/hip_runtime.h>

typedef __attribute__((ext_vector_type(16))) _Float16 v16h;
typedef __attribute__((ext_vector_type(8)))  _Float16 v8h;
typedef __attribute__((ext_vector_type(8)))  float    v8f;
typedef __attribute__((ext_vector_type(4)))  float    v4f;

#define T_STEPS 512
#define BATCH   128
#define DIM     256
#define HID     256

static __device__ __forceinline__ v16h cat8(v8h lo, v8h hi) {
    return __builtin_shufflevector(lo, hi, 0,1,2,3,4,5,6,7,8,9,10,11,12,13,14,15);
}

// swizzled LDS offset for h element (b, n): 8-half blocks rotated by row
static __device__ __forceinline__ int hswz(int b, int n) {
    return (b << 8) + ((((n >> 3) + b) & 31) << 3) + (n & 7);
}

// ---------------------------------------------------------------------------
// Kernel 0: convert Wi [H,D] and Wh [H,H] (f32) into f16 in workspace.
// w16[0 .. 65535]      = Wi16  (row-major [H][D])
// w16[65536 .. 131071] = Wh16  (row-major [H][H])
// ---------------------------------------------------------------------------
__global__ __launch_bounds__(256) void convert_weights(
    const float* __restrict__ Wi, const float* __restrict__ Wh,
    _Float16* __restrict__ w16)
{
    int i = blockIdx.x * 256 + threadIdx.x;   // 0 .. 131071
    float v = (i < HID * DIM) ? Wi[i] : Wh[i - HID * DIM];
    w16[i] = (_Float16)v;
}

// ---------------------------------------------------------------------------
// Kernel 1: P[r][n] = X[r][:]·Wi[n][:] + bh[n]  for r in [0, T*B), written to
// d_out (pre-activations; kernel 2 overwrites with sigmoid(P + h·Wh^T)).
// One wave computes a 16x64 strip: 4 WMMA tiles, K looped in chunks of 32.
// ---------------------------------------------------------------------------
__global__ __launch_bounds__(256) void precompute_xWi(
    const float* __restrict__ X, const _Float16* __restrict__ Wi16,
    const float* __restrict__ bh, float* __restrict__ out)
{
    const int lane = threadIdx.x & 31;
    const int wv   = threadIdx.x >> 5;
    const int gw   = blockIdx.x * 8 + wv;     // global wave-tile id
    const int m_tile = gw >> 2;               // 0 .. 4095  (rows of T*B, /16)
    const int n0     = (gw & 3) * 64;         // 0,64,128,192
    const int l16    = lane & 15;
    const int hi     = lane >> 4;

    const float* Xrow = X + (size_t)(m_tile * 16 + l16) * DIM;

    v8f acc[4] = {};
    #pragma unroll
    for (int k0 = 0; k0 < DIM; k0 += 32) {
        const int base = k0 + 8 * hi;
        // A fragment: X row (f32 -> f16 on the fly)
        v4f f0 = *(const v4f*)(Xrow + base);
        v4f f1 = *(const v4f*)(Xrow + base + 4);
        v4f f2 = *(const v4f*)(Xrow + base + 16);
        v4f f3 = *(const v4f*)(Xrow + base + 20);
        v16h a;
        #pragma unroll
        for (int i = 0; i < 4; ++i) {
            a[i]      = (_Float16)f0[i];
            a[4 + i]  = (_Float16)f1[i];
            a[8 + i]  = (_Float16)f2[i];
            a[12 + i] = (_Float16)f3[i];
        }
        #pragma unroll
        for (int t4 = 0; t4 < 4; ++t4) {
            const _Float16* wrow =
                Wi16 + (size_t)(n0 + t4 * 16 + l16) * DIM + base;
            v16h b = cat8(*(const v8h*)wrow, *(const v8h*)(wrow + 16));
            acc[t4] = __builtin_amdgcn_wmma_f32_16x16x32_f16(
                false, a, false, b, (short)0, acc[t4], false, false);
        }
    }
    // epilogue: add bias, store f32 pre-activations
    #pragma unroll
    for (int t4 = 0; t4 < 4; ++t4) {
        const int n = n0 + t4 * 16 + l16;
        const float bias = bh[n];
        size_t rowbase = (size_t)(m_tile * 16 + 8 * hi) * HID + n;
        #pragma unroll
        for (int v = 0; v < 8; ++v)
            out[rowbase + (size_t)v * HID] = acc[t4][v] + bias;
    }
}

// ---------------------------------------------------------------------------
// Kernel 2: persistent single-workgroup recurrence on one WGP.
// h (f16) in double-buffered LDS (2 x 64 KB, block-swizzled): read buf[t&1],
// write buf[(t+1)&1] -> only ONE workgroup barrier per step.
// Each of 32 waves owns a 2x2 tile group (32 rows x 32 cols): per K-chunk
// 2 A-frags from LDS + 2 B-frags from L2-resident Wh16 + 4 WMMAs, balancing
// LDS and vmem pipe traffic (~512 KB/step each).
// ---------------------------------------------------------------------------
__global__ __launch_bounds__(1024) void rnn_recurrent(
    const float* __restrict__ h0, const _Float16* __restrict__ Wh16,
    float* __restrict__ out)
{
    __shared__ _Float16 hbuf[2][BATCH * HID];   // 131072 bytes total

    const int tid  = threadIdx.x;
    const int lane = tid & 31;
    const int wv   = tid >> 5;                // 0..31
    const int l16  = lane & 15;
    const int hi   = lane >> 4;
    const int gm   = (wv & 3) * 32;           // first batch row of 2 m-tiles
    const int gn   = (wv >> 2) * 32;          // first hidden col of 2 n-tiles

    // init h buffer 0 from h0 (f32) with swizzle
    for (int i = tid; i < BATCH * HID; i += 1024) {
        int b = i >> 8, n = i & 255;
        hbuf[0][hswz(b, n)] = (_Float16)h0[i];
    }
    __syncthreads();

    const int arow0 = gm + l16;
    const int arow1 = arow0 + 16;

    for (int t = 0; t < T_STEPS; ++t) {
        const _Float16* __restrict__ hr = hbuf[t & 1];
        _Float16* __restrict__       hw = hbuf[(t + 1) & 1];

        v8f acc[2][2] = {};
        #pragma unroll
        for (int k0 = 0; k0 < HID; k0 += 32) {
            const int kb = (k0 >> 3) + hi;
            v16h a0 = cat8(
                *(const v8h*)&hr[(arow0 << 8) + (((kb + arow0) & 31) << 3)],
                *(const v8h*)&hr[(arow0 << 8) + (((kb + 2 + arow0) & 31) << 3)]);
            v16h a1 = cat8(
                *(const v8h*)&hr[(arow1 << 8) + (((kb + arow1) & 31) << 3)],
                *(const v8h*)&hr[(arow1 << 8) + (((kb + 2 + arow1) & 31) << 3)]);
            const int base = k0 + 8 * hi;
            #pragma unroll
            for (int ni = 0; ni < 2; ++ni) {
                const _Float16* wrow =
                    Wh16 + (size_t)(gn + ni * 16 + l16) * HID + base;
                v16h b = cat8(*(const v8h*)wrow, *(const v8h*)(wrow + 16));
                acc[0][ni] = __builtin_amdgcn_wmma_f32_16x16x32_f16(
                    false, a0, false, b, (short)0, acc[0][ni], false, false);
                acc[1][ni] = __builtin_amdgcn_wmma_f32_16x16x32_f16(
                    false, a1, false, b, (short)0, acc[1][ni], false, false);
            }
        }

        // epilogue: sigmoid(acc + P[t]) -> d_out and the other LDS buffer
        float* outT = out + (size_t)t * (BATCH * HID);
        #pragma unroll
        for (int mi = 0; mi < 2; ++mi) {
            #pragma unroll
            for (int ni = 0; ni < 2; ++ni) {
                const int n = gn + ni * 16 + l16;
                const int brow = gm + mi * 16 + 8 * hi;
                #pragma unroll
                for (int v = 0; v < 8; ++v) {
                    const int b = brow + v;
                    const size_t idx = (size_t)b * HID + n;
                    float x = acc[mi][ni][v] + outT[idx];   // + P[t]
                    float s = __builtin_amdgcn_rcpf(1.0f + __expf(-x));
                    outT[idx] = s;
                    hw[hswz(b, n)] = (_Float16)s;
                }
            }
        }
        __syncthreads();   // h_new published; next step reads hw
    }
}

extern "C" void kernel_launch(void* const* d_in, const int* in_sizes, int n_in,
                              void* d_out, int out_size, void* d_ws, size_t ws_size,
                              hipStream_t stream) {
    (void)in_sizes; (void)n_in; (void)out_size; (void)ws_size;
    const float* X  = (const float*)d_in[0];
    const float* h0 = (const float*)d_in[1];
    const float* Wi = (const float*)d_in[2];
    const float* Wh = (const float*)d_in[3];
    const float* bh = (const float*)d_in[4];
    float* out = (float*)d_out;
    _Float16* w16 = (_Float16*)d_ws;          // 256 KB: Wi16 then Wh16

    convert_weights<<<(2 * HID * DIM) / 256, 256, 0, stream>>>(Wi, Wh, w16);
    // (T*B/16) m-tiles * 4 n-groups = 16384 wave-tiles / 8 waves per block
    precompute_xWi<<<2048, 256, 0, stream>>>(X, w16, bh, out);
    rnn_recurrent<<<1, 1024, 0, stream>>>(h0, w16 + HID * DIM, out);
}
